// GAT_Stu_38405597560838
// MI455X (gfx1250) — matrix operand order repaired and checked by reading, not compiled, not run on hardware
//
#include <hip/hip_runtime.h>
#include <hip/hip_bf16.h>
#include <math.h>

#define N_NODES   20000
#define N_EDGES   320000
#define E_TOT     (N_EDGES + N_NODES)   // +self loops = 340000
#define NFEAT     256
#define NHID      32
#define HEADS     8
#define HD        (HEADS * NHID)        // 256
#define NCLASS    40
#define NCLASS_PAD 48
#define NEG_SLOPE 0.2f

typedef __bf16 bf16_t;
typedef bf16_t v16bf __attribute__((ext_vector_type(16)));
typedef float  v8f   __attribute__((ext_vector_type(8)));
typedef unsigned short u16;
typedef u16 u16x8 __attribute__((ext_vector_type(8)));

union ABFrag { u16x8 h[2]; v16bf v; };   // 32 bytes both views

__device__ __forceinline__ u16 f2bf(float f) {
    union { bf16_t b; u16 u; } c; c.b = (bf16_t)f; return c.u;
}

// ---------- CDNA5 atomic helpers ----------
__device__ __forceinline__ void atomic_max_f32_dev(float* addr, float val) {
    // GLOBAL_ATOMIC_MAX_NUM_F32, no-return (TH[0]=0), device scope
    asm volatile("global_atomic_max_num_f32 %0, %1, off scope:SCOPE_DEV"
                 :: "v"(addr), "v"(val) : "memory");
}
__device__ __forceinline__ void atomic_add_f32_dev(float* addr, float val) {
    (void)__hip_atomic_fetch_add(addr, val, __ATOMIC_RELAXED, __HIP_MEMORY_SCOPE_AGENT);
}

// ---------- fill ----------
__global__ void gat_fill_f32(float* __restrict__ p, float v, int n) {
    int i = blockIdx.x * blockDim.x + threadIdx.x;
    if (i < n) p[i] = v;
}

// ---------- operand conversion kernels (run once per call) ----------
__global__ void gat_cvt_x(const float* __restrict__ x, u16* __restrict__ xb, int n) {
    int i = blockIdx.x * blockDim.x + threadIdx.x;
    if (i >= n) return;
    float v = x[i];
    v = (v == v) ? v : 0.0f;                 // NaN imputation
    xb[i] = f2bf(v);
}
__global__ void gat_cvt_relu(const float* __restrict__ h, u16* __restrict__ hb, int n) {
    int i = blockIdx.x * blockDim.x + threadIdx.x;
    if (i >= n) return;
    float v = h[i];
    hb[i] = f2bf(v > 0.0f ? v : 0.0f);
}
// W1 [256,256] -> W1^T bf16 [256,256]
__global__ void gat_cvt_w1t(const float* __restrict__ W1, u16* __restrict__ w1t) {
    int t = blockIdx.x * blockDim.x + threadIdx.x;
    if (t >= HD * NFEAT) return;
    int n = t >> 8, k = t & 255;
    w1t[(size_t)n * NFEAT + k] = f2bf(W1[(size_t)k * HD + n]);
}
// W2 [256,40] -> padded W2^T bf16 [48,256]
__global__ void gat_cvt_w2t(const float* __restrict__ W2, u16* __restrict__ w2t) {
    int t = blockIdx.x * blockDim.x + threadIdx.x;
    if (t >= NCLASS_PAD * HD) return;
    int n = t >> 8, k = t & 255;
    w2t[t] = (n < NCLASS) ? f2bf(W2[(size_t)k * NCLASS + n]) : (u16)0;
}

// ---------- panel GEMM: C[m,:] = A(bf16 [M,256]) @ BT(bf16 [cols,256])^T ----------
// Block: 8 waves, each wave one 16-row M-tile, all waves share NT 16-col N-tiles.
// B panel (NT*16 cols x K=256 bf16) staged in LDS via async-to-LDS copies.
#define BT_PITCH 264   // 256 + 8 ushort pad -> 528B col stride, conflict-spread
template<int NT>
__global__ __launch_bounds__(256)
void gat_gemm_panel(const u16* __restrict__ A, const u16* __restrict__ BT,
                    float* __restrict__ C, int m_tiles, int out_ld, int out_ncols) {
    __shared__ u16 bt[NT * 16 * BT_PITCH];
    const int ncols   = NT * 16;
    const int colbase = blockIdx.y * ncols;

    // ---- async cooperative load of B panel into LDS (ASYNCcnt path) ----
    for (int chunk = threadIdx.x; chunk < ncols * 32; chunk += 256) {
        int col = chunk >> 5;           // column within panel
        int kc  = chunk & 31;           // 16B chunk along K
        const u16* gsrc = BT + (size_t)(colbase + col) * NFEAT + kc * 8;
        unsigned lds = (unsigned)(uintptr_t)&bt[col * BT_PITCH + kc * 8];
        asm volatile("global_load_async_to_lds_b128 %0, %1, off"
                     :: "v"(lds), "v"(gsrc) : "memory");
    }
    asm volatile("s_wait_asynccnt 0" ::: "memory");
    __syncthreads();

    const int lane = threadIdx.x & 31;
    const int wave = threadIdx.x >> 5;
    const int tm   = blockIdx.x * 8 + wave;
    if (tm >= m_tiles) return;                // wave-uniform: EXEC stays all-ones
    const int row = lane & 15;
    const int kh  = lane >> 4;                // K-half select

    v8f acc[NT];
#pragma unroll
    for (int t = 0; t < NT; ++t) acc[t] = {};

    const u16* arow = A + (size_t)(tm * 16 + row) * NFEAT;
    for (int k0 = 0; k0 < NFEAT; k0 += 32) {
        ABFrag a;
        // A 16x32 bf16 layout: e<8 -> K=k0+kh*8+e ; e>=8 -> K=k0+16+kh*8+(e-8)
        a.h[0] = *(const u16x8*)(arow + k0 + kh * 8);
        a.h[1] = *(const u16x8*)(arow + k0 + 16 + kh * 8);
#pragma unroll
        for (int t = 0; t < NT; ++t) {
            ABFrag b;
            // B 32x16 bf16 layout: column (t*16+row) per lane, e<8 -> K=k0+kh*8+e ; e>=8 -> K=k0+16+kh*8+(e-8)
            const u16* bp = &bt[(t * 16 + row) * BT_PITCH + k0 + kh * 8];
            b.h[0] = *(const u16x8*)bp;        // ds_load_b128
            b.h[1] = *(const u16x8*)(bp + 16);
            acc[t] = __builtin_amdgcn_wmma_f32_16x16x32_bf16(false, a.v, false, b.v,
                                                             (short)0, acc[t], false, false);
        }
    }
#pragma unroll
    for (int t = 0; t < NT; ++t)
#pragma unroll
        for (int r = 0; r < 8; ++r) {
            int mo = tm * 16 + kh * 8 + r;    // D: lanes 16-31 hold M=8..15
            int no = colbase + t * 16 + (lane & 15);
            if (no < out_ncols)
                C[(size_t)mo * out_ld + no] = acc[t][r];
        }
}

// ---------- per-node attention logits, layer 1 ----------
__global__ void gat_logits1(const float* __restrict__ h1pre,
                            const float* __restrict__ a_src,
                            const float* __restrict__ a_dst,
                            float* __restrict__ als, float* __restrict__ ald) {
    int t = blockIdx.x * blockDim.x + threadIdx.x;
    if (t >= N_NODES * HEADS) return;
    int nd = t >> 3, hh = t & 7;
    const float* hp = h1pre + (size_t)nd * HD + hh * NHID;
    const float* as = a_src + hh * NHID;
    const float* ad = a_dst + hh * NHID;
    float s1 = 0.f, s2 = 0.f;
#pragma unroll
    for (int c = 0; c < NHID; ++c) { float v = hp[c]; s1 += v * as[c]; s2 += v * ad[c]; }
    als[t] = s1; ald[t] = s2;
}

// ---------- per-node attention logits, layer 2 (1 head) ----------
__global__ void gat_logits2(const float* __restrict__ h2pre,
                            const float* __restrict__ a_src,
                            const float* __restrict__ a_dst,
                            float* __restrict__ als, float* __restrict__ ald) {
    int nd = blockIdx.x * blockDim.x + threadIdx.x;
    if (nd >= N_NODES) return;
    const float* hp = h2pre + (size_t)nd * NCLASS;
    float s1 = 0.f, s2 = 0.f;
#pragma unroll
    for (int c = 0; c < NCLASS; ++c) { float v = hp[c]; s1 += v * a_src[c]; s2 += v * a_dst[c]; }
    als[nd] = s1; ald[nd] = s2;
}

__device__ __forceinline__ void edge_sd(int e, const int* __restrict__ ei, int& s, int& d) {
    if (e < N_EDGES) { s = ei[e]; d = ei[N_EDGES + e]; }
    else             { s = d = e - N_EDGES; }            // self loops
}

// ---------- layer 1 edge passes (per edge*head) ----------
__global__ void gat_edge1_max(const int* __restrict__ ei, const float* __restrict__ als,
                              const float* __restrict__ ald, float* __restrict__ m1) {
    int t = blockIdx.x * blockDim.x + threadIdx.x;
    if (t >= E_TOT * HEADS) return;
    int e = t >> 3, hh = t & 7, s, d;
    edge_sd(e, ei, s, d);
    float v = als[s * HEADS + hh] + ald[d * HEADS + hh];
    v = v > 0.f ? v : NEG_SLOPE * v;
    atomic_max_f32_dev(&m1[d * HEADS + hh], v);
}

__global__ void gat_edge1_sum(const int* __restrict__ ei, const float* __restrict__ als,
                              const float* __restrict__ ald, const float* __restrict__ m1,
                              float* __restrict__ den1) {
    int t = blockIdx.x * blockDim.x + threadIdx.x;
    if (t >= E_TOT * HEADS) return;
    int e = t >> 3, hh = t & 7, s, d;
    edge_sd(e, ei, s, d);
    float v = als[s * HEADS + hh] + ald[d * HEADS + hh];
    v = v > 0.f ? v : NEG_SLOPE * v;
    atomic_add_f32_dev(&den1[d * HEADS + hh], __expf(v - m1[d * HEADS + hh]));
}

__global__ void gat_edge1_scatter(const int* __restrict__ ei, const float* __restrict__ als,
                                  const float* __restrict__ ald, const float* __restrict__ m1,
                                  const float* __restrict__ den1, const float* __restrict__ h1pre,
                                  float* __restrict__ h1out) {
    int t = blockIdx.x * blockDim.x + threadIdx.x;
    if (t >= E_TOT * HEADS) return;
    int e = t >> 3, hh = t & 7, s, d;
    edge_sd(e, ei, s, d);
    int di = d * HEADS + hh;
    float v = als[s * HEADS + hh] + ald[di];
    v = v > 0.f ? v : NEG_SLOPE * v;
    float alpha = __expf(v - m1[di]) / den1[di];
    const float* hs = h1pre + (size_t)s * HD + hh * NHID;
    float* od = h1out + (size_t)d * HD + hh * NHID;
#pragma unroll
    for (int c = 0; c < NHID; ++c)
        atomic_add_f32_dev(&od[c], hs[c] * alpha);
}

// ---------- layer 2 edge passes (per edge, 1 head) ----------
__global__ void gat_edge2_max(const int* __restrict__ ei, const float* __restrict__ als,
                              const float* __restrict__ ald, float* __restrict__ m2) {
    int e = blockIdx.x * blockDim.x + threadIdx.x;
    if (e >= E_TOT) return;
    int s, d; edge_sd(e, ei, s, d);
    float v = als[s] + ald[d];
    v = v > 0.f ? v : NEG_SLOPE * v;
    atomic_max_f32_dev(&m2[d], v);
}

__global__ void gat_edge2_sum(const int* __restrict__ ei, const float* __restrict__ als,
                              const float* __restrict__ ald, const float* __restrict__ m2,
                              float* __restrict__ den2) {
    int e = blockIdx.x * blockDim.x + threadIdx.x;
    if (e >= E_TOT) return;
    int s, d; edge_sd(e, ei, s, d);
    float v = als[s] + ald[d];
    v = v > 0.f ? v : NEG_SLOPE * v;
    atomic_add_f32_dev(&den2[d], __expf(v - m2[d]));
}

__global__ void gat_edge2_scatter(const int* __restrict__ ei, const float* __restrict__ als,
                                  const float* __restrict__ ald, const float* __restrict__ m2,
                                  const float* __restrict__ den2, const float* __restrict__ h2pre,
                                  float* __restrict__ out2) {
    int e = blockIdx.x * blockDim.x + threadIdx.x;
    if (e >= E_TOT) return;
    int s, d; edge_sd(e, ei, s, d);
    float v = als[s] + ald[d];
    v = v > 0.f ? v : NEG_SLOPE * v;
    float alpha = __expf(v - m2[d]) / den2[d];
    const float* hs = h2pre + (size_t)s * NCLASS;
    float* od = out2 + (size_t)d * NCLASS;
#pragma unroll
    for (int c = 0; c < NCLASS; ++c)
        atomic_add_f32_dev(&od[c], hs[c] * alpha);
}

// ---------- bias broadcast add ----------
__global__ void gat_add_bias(float* __restrict__ out, const float* __restrict__ b,
                             int n, int stride) {
    int t = blockIdx.x * blockDim.x + threadIdx.x;
    if (t >= n) return;
    out[t] += b[t % stride];
}

// =====================================================================
extern "C" void kernel_launch(void* const* d_in, const int* in_sizes, int n_in,
                              void* d_out, int out_size, void* d_ws, size_t ws_size,
                              hipStream_t stream) {
    const float* x      = (const float*)d_in[0];
    const int*   ei     = (const int*)  d_in[1];
    const float* W1     = (const float*)d_in[2];
    const float* a_src1 = (const float*)d_in[3];
    const float* a_dst1 = (const float*)d_in[4];
    const float* b1     = (const float*)d_in[5];
    const float* W2     = (const float*)d_in[6];
    const float* a_src2 = (const float*)d_in[7];
    const float* a_dst2 = (const float*)d_in[8];
    const float* b2     = (const float*)d_in[9];

    float* out2  = (float*)d_out;                             // h2: [20000,40]
    float* h1out = (float*)d_out + (size_t)N_NODES * NCLASS;  // h1: [20000,256]

    // workspace layout
    float* ws    = (float*)d_ws;
    float* h1pre = ws;                                    // 5,120,000 f32
    float* als1  = h1pre + (size_t)N_NODES * HD;          // 160,000
    float* ald1  = als1 + N_NODES * HEADS;
    float* m1    = ald1 + N_NODES * HEADS;
    float* den1  = m1   + N_NODES * HEADS;
    float* h2pre = den1 + N_NODES * HEADS;                // 800,000
    float* als2  = h2pre + (size_t)N_NODES * NCLASS;      // 20,000
    float* ald2  = als2 + N_NODES;
    float* m2    = ald2 + N_NODES;
    float* den2  = m2   + N_NODES;
    u16*   xb    = (u16*)(den2 + N_NODES);                // bf16 x / later relu(h1): 5,120,000 u16
    u16*   h1b   = xb;                                    // aliases xb (x dead after GEMM1)
    u16*   w1t   = xb + (size_t)N_NODES * NFEAT;          // 65,536 u16
    u16*   w2t   = w1t + (size_t)HD * NFEAT;              // 12,288 u16

    const int B = 256;
    auto cdiv = [](int a, int b) { return (a + b - 1) / b; };

    // init: zero outputs/accumulators, -inf for segment maxes
    gat_fill_f32<<<cdiv(out_size, B), B, 0, stream>>>((float*)d_out, 0.0f, out_size);
    gat_fill_f32<<<cdiv(N_NODES * HEADS, B), B, 0, stream>>>(m1, -INFINITY, N_NODES * HEADS);
    gat_fill_f32<<<cdiv(N_NODES * HEADS, B), B, 0, stream>>>(den1, 0.0f, N_NODES * HEADS);
    gat_fill_f32<<<cdiv(N_NODES, B), B, 0, stream>>>(m2, -INFINITY, N_NODES);
    gat_fill_f32<<<cdiv(N_NODES, B), B, 0, stream>>>(den2, 0.0f, N_NODES);

    // operand conversion (bf16, transposed weights)
    gat_cvt_x  <<<cdiv(N_NODES * NFEAT, B), B, 0, stream>>>(x, xb, N_NODES * NFEAT);
    gat_cvt_w1t<<<cdiv(HD * NFEAT, B), B, 0, stream>>>(W1, w1t);
    gat_cvt_w2t<<<cdiv(NCLASS_PAD * HD, B), B, 0, stream>>>(W2, w2t);

    const int MT = N_NODES / 16;     // 1250 M-tiles
    // ----- layer 1 -----
    gat_gemm_panel<4><<<dim3(cdiv(MT, 8), HD / 64), B, 0, stream>>>(xb, w1t, h1pre, MT, HD, HD);
    gat_logits1<<<cdiv(N_NODES * HEADS, B), B, 0, stream>>>(h1pre, a_src1, a_dst1, als1, ald1);
    int eg1 = cdiv(E_TOT * HEADS, B);
    gat_edge1_max    <<<eg1, B, 0, stream>>>(ei, als1, ald1, m1);
    gat_edge1_sum    <<<eg1, B, 0, stream>>>(ei, als1, ald1, m1, den1);
    gat_edge1_scatter<<<eg1, B, 0, stream>>>(ei, als1, ald1, m1, den1, h1pre, h1out);
    gat_add_bias<<<cdiv(N_NODES * HD, B), B, 0, stream>>>(h1out, b1, N_NODES * HD, HD);

    // ----- layer 2 -----
    gat_cvt_relu<<<cdiv(N_NODES * HD, B), B, 0, stream>>>(h1out, h1b, N_NODES * HD);
    gat_gemm_panel<3><<<dim3(cdiv(MT, 8), 1), B, 0, stream>>>(h1b, w2t, h2pre, MT, NCLASS, NCLASS);
    gat_logits2<<<cdiv(N_NODES, B), B, 0, stream>>>(h2pre, a_src2, a_dst2, als2, ald2);
    int eg2 = cdiv(E_TOT, B);
    gat_edge2_max    <<<eg2, B, 0, stream>>>(ei, als2, ald2, m2);
    gat_edge2_sum    <<<eg2, B, 0, stream>>>(ei, als2, ald2, m2, den2);
    gat_edge2_scatter<<<eg2, B, 0, stream>>>(ei, als2, ald2, m2, den2, h2pre, out2);
    gat_add_bias<<<cdiv(N_NODES * NCLASS, B), B, 0, stream>>>(out2, b2, N_NODES * NCLASS, NCLASS);
}
